// Net_37056977829968
// MI455X (gfx1250) — matrix operand order repaired and checked by reading, not compiled, not run on hardware
//
#include <hip/hip_runtime.h>
#include <math.h>

// ---------------------------------------------------------------------------
// Types for CDNA5 WMMA (wave32): v16bf = 8 VGPRs of packed bf16, v8f = 8 VGPRs
// ---------------------------------------------------------------------------
typedef __attribute__((ext_vector_type(16))) __bf16 v16bf;
typedef __attribute__((ext_vector_type(8)))  float  v8f;

#define NEG_SLOPE 0.2f
#define EPS_F 1e-16f

// ---------------------------------------------------------------------------
// Helpers
// ---------------------------------------------------------------------------
__device__ __forceinline__ void atomicMaxFloat(float* addr, float value) {
  // monotonic-bit-pattern trick; addr must be initialized to -inf
  if (value >= 0.f)
    atomicMax((int*)addr, __float_as_int(value));
  else
    atomicMin((unsigned int*)addr, __float_as_uint(value));
}

__global__ void fill_f32(float* __restrict__ p, float v, size_t n) {
  size_t i = (size_t)blockIdx.x * blockDim.x + threadIdx.x;
  size_t stride = (size_t)gridDim.x * blockDim.x;
  for (; i < n; i += stride) p[i] = v;
}

// ---------------------------------------------------------------------------
// Pack a row-major fp32 weight matrix W[K, Ncols] into the WMMA B-operand
// layout as bf16.  For each 32x16 (KxN) tile, lane L holds column n = nt*16 +
// (L&15); lanes 0-15 hold K = kt*32 + {0..7,16..23}, lanes 16-31 hold K = kt*32 +
// {8..15,24..31}; elements 0-7 and 8-15 are two K-contiguous runs of 8.
// wb element index = ((kt*ntiles + nt)*32 + lane)*16 + j
// ---------------------------------------------------------------------------
__global__ void pack_w_bf16(const float* __restrict__ W, int ldw, int Klim,
                            int ktiles, int ntiles, __bf16* __restrict__ wb) {
  int t = blockIdx.x * blockDim.x + threadIdx.x;
  int total = ktiles * ntiles * 32 * 16;
  if (t >= total) return;
  int j    = t & 15;
  int lane = (t >> 4) & 31;
  int tile = t >> 9;                 // kt*ntiles + nt
  int nt   = tile % ntiles;
  int kt   = tile / ntiles;
  int K = kt * 32 + ((lane >= 16) ? 8 : 0) + j + ((j >= 8) ? 8 : 0);
  int n = nt * 16 + (lane & 15);
  float v = (K < Klim) ? W[(size_t)K * ldw + n] : 0.f;
  wb[t] = (__bf16)v;
}

// ---------------------------------------------------------------------------
// WMMA GEMM with async-DMA double-buffered B panels in LDS.
//   out[Nrows, NT*16] = in[Nrows, KLIM] (fp32 -> bf16 in regs) x packed-B.
// One wave computes a 16-row x (NT*16)-col strip.  The 8-wave workgroup
// shares one K-panel of B (NT*1KB) staged into LDS with
// global_load_async_to_lds_b128 one iteration ahead (ASYNCcnt pipeline),
// consumed via ds_load_b128.  A-tail (K % 32 != 0) is a uniform branch taken
// only on the final iteration.
// ---------------------------------------------------------------------------
template <int NT, int KLIM, int KTILES, int INSTRIDE>
__global__ void __launch_bounds__(256)
wmma_gemm_bf16(const float* __restrict__ in,
               const unsigned int* __restrict__ wb,
               float* __restrict__ out, int out_stride, int Nrows) {
  constexpr int KTFULL     = KLIM / 32;            // tiles with no K guard
  constexpr int PANEL_U32  = NT * 32 * 8;          // uints per K-panel
  constexpr int PANEL_B    = PANEL_U32 * 4;        // bytes per K-panel
  constexpr int NCHUNK     = PANEL_B / 16;         // b128 chunks per panel
  __shared__ unsigned int sB[2 * PANEL_U32];

  const int tid  = threadIdx.x;
  const int lane = tid & 31;
  const int wave = (blockIdx.x * blockDim.x + tid) >> 5;
  int m0 = wave * 16;
  if (m0 > Nrows - 16) m0 = Nrows - 16;  // clamp: keep all waves in barriers
  const int hi = lane >> 4;
  const float* xrow = in + (size_t)(m0 + (lane & 15)) * INSTRIDE;

  // async-stage one K-panel of packed B into LDS buffer `buf`
  auto stage = [&](int kt, int buf) {
    const unsigned long long gbase =
        (unsigned long long)(size_t)wb + (unsigned long long)kt * PANEL_B;
    const unsigned lbase = (unsigned)(size_t)(&sB[buf * PANEL_U32]);
#pragma unroll
    for (int c0 = 0; c0 < NCHUNK; c0 += 256) {
      int c = c0 + tid;
      if (c < NCHUNK) {
        unsigned loff = lbase + (unsigned)c * 16u;
        unsigned long long gaddr = gbase + (unsigned long long)c * 16u;
        asm volatile("global_load_async_to_lds_b128 %0, %1, off"
                     :: "v"(loff), "v"(gaddr)
                     : "memory");
      }
    }
  };

  stage(0, 0);
  asm volatile("s_wait_asynccnt 0x0" ::: "memory");
  __syncthreads();

  v8f acc[NT];
#pragma unroll
  for (int i = 0; i < NT; ++i)
#pragma unroll
    for (int j = 0; j < 8; ++j) acc[i][j] = 0.f;

  for (int kt = 0; kt < KTILES; ++kt) {
    const int cur = kt & 1;
    if (kt + 1 < KTILES) stage(kt + 1, cur ^ 1);  // prefetch next panel

    // ---- A fragment: fp32 -> bf16 in registers ----
    // A layout (16-bit A 16x32): row = lane&15; lanes<16 hold K {0..7,16..23},
    // lanes>=16 hold K {8..15,24..31}.
    const int kbase = kt * 32 + hi * 8;
    v16bf A;
    if (kt < KTFULL) {  // uniform branch; tail only on last iteration
      float4 f0 = *(const float4*)(xrow + kbase);
      float4 f1 = *(const float4*)(xrow + kbase + 4);
      float4 f2 = *(const float4*)(xrow + kbase + 16);
      float4 f3 = *(const float4*)(xrow + kbase + 20);
      A[0]  = (__bf16)f0.x; A[1]  = (__bf16)f0.y; A[2]  = (__bf16)f0.z; A[3]  = (__bf16)f0.w;
      A[4]  = (__bf16)f1.x; A[5]  = (__bf16)f1.y; A[6]  = (__bf16)f1.z; A[7]  = (__bf16)f1.w;
      A[8]  = (__bf16)f2.x; A[9]  = (__bf16)f2.y; A[10] = (__bf16)f2.z; A[11] = (__bf16)f2.w;
      A[12] = (__bf16)f3.x; A[13] = (__bf16)f3.y; A[14] = (__bf16)f3.z; A[15] = (__bf16)f3.w;
    } else {
#pragma unroll
      for (int j = 0; j < 16; ++j) {
        int K = kbase + j + ((j >= 8) ? 8 : 0);
        A[j] = (K < KLIM) ? (__bf16)xrow[K] : (__bf16)0.f;
      }
    }

    // ---- B tiles from LDS, batched loads then back-to-back WMMAs ----
    const unsigned int* bpan = &sB[cur * PANEL_U32];
    union { uint4 q[2]; v16bf v; } B[NT];
#pragma unroll
    for (int nt = 0; nt < NT; ++nt) {
      const uint4* bp = (const uint4*)(bpan + ((size_t)(nt * 32 + lane) * 8));
      B[nt].q[0] = bp[0];
      B[nt].q[1] = bp[1];
    }
#pragma unroll
    for (int nt = 0; nt < NT; ++nt)
      acc[nt] = __builtin_amdgcn_wmma_f32_16x16x32_bf16(
          false, A, false, B[nt].v, (short)0, acc[nt], false, false);

    asm volatile("s_wait_asynccnt 0x0" ::: "memory");  // next panel landed
    __syncthreads();  // all waves done reading `cur`; staged data visible
  }

  // C/D layout: VGPR r, lanes<16 -> row m0+r, lanes>=16 -> row m0+8+r; col = lane&15
#pragma unroll
  for (int nt = 0; nt < NT; ++nt)
#pragma unroll
    for (int r = 0; r < 8; ++r)
      out[(size_t)(m0 + r + hi * 8) * out_stride + nt * 16 + (lane & 15)] =
          acc[nt][r];
}

// ---------------------------------------------------------------------------
// Per-node attention scalars: s[k][n*H+h] = h[n,h,:] . a[k,h,:]   (C = 16)
// ---------------------------------------------------------------------------
__global__ void node_logits8(const float* __restrict__ h,
                             const float* __restrict__ a,
                             float* __restrict__ s, int N) {
  int t = blockIdx.x * blockDim.x + threadIdx.x;  // t = n*8 + head
  if (t >= N * 8) return;
  int hh = t & 7;
  const float* hv = h + (size_t)(t >> 3) * 128 + hh * 16;
  size_t NH = (size_t)N * 8;
#pragma unroll
  for (int k = 0; k < 3; ++k) {
    const float* av = a + k * 128 + hh * 16;
    float acc = 0.f;
#pragma unroll
    for (int c = 0; c < 16; ++c) acc += hv[c] * av[c];
    s[k * NH + t] = acc;
  }
}

__global__ void node_logits1(const float* __restrict__ h2,
                             const float* __restrict__ a2,
                             float* __restrict__ s, int N) {
  int n = blockIdx.x * blockDim.x + threadIdx.x;
  if (n >= N) return;
  const float* hv = h2 + (size_t)n * 16;
#pragma unroll
  for (int k = 0; k < 3; ++k) {
    float acc = 0.f;
#pragma unroll
    for (int c = 0; c < 16; ++c) acc += hv[c] * a2[k * 16 + c];
    s[(size_t)k * N + n] = acc;
  }
}

// ---------------------------------------------------------------------------
// Segment-softmax pass 1: running max per (dest node, head)
// ---------------------------------------------------------------------------
template <int H>
__global__ void path_max(const int* __restrict__ pi, int P,
                         const float* __restrict__ s, int N,
                         float* __restrict__ mbuf) {
  int t = blockIdx.x * blockDim.x + threadIdx.x;
  if (t >= P * H) return;
  int p = t / H, hh = t - p * H;
  int hd = pi[p], md = pi[P + p], tl = pi[2 * P + p];
  size_t NH = (size_t)N * H;
  float e = s[(size_t)hd * H + hh] + s[NH + (size_t)md * H + hh] +
            s[2 * NH + (size_t)tl * H + hh];
  e = (e > 0.f) ? e : NEG_SLOPE * e;
  atomicMaxFloat(&mbuf[(size_t)hd * H + hh], e);
}

// pass 2: sum of exp(e - max) per (dest node, head)
template <int H>
__global__ void path_sumexp(const int* __restrict__ pi, int P,
                            const float* __restrict__ s, int N,
                            const float* __restrict__ mbuf,
                            float* __restrict__ ssum) {
  int t = blockIdx.x * blockDim.x + threadIdx.x;
  if (t >= P * H) return;
  int p = t / H, hh = t - p * H;
  int hd = pi[p], md = pi[P + p], tl = pi[2 * P + p];
  size_t NH = (size_t)N * H;
  float e = s[(size_t)hd * H + hh] + s[NH + (size_t)md * H + hh] +
            s[2 * NH + (size_t)tl * H + hh];
  e = (e > 0.f) ? e : NEG_SLOPE * e;
  float ex = expf(e - mbuf[(size_t)hd * H + hh]);
  atomicAdd(&ssum[(size_t)hd * H + hh], ex);
}

// ---------------------------------------------------------------------------
// Layer-1 aggregation: one wave32 per path.  Lanes 0..7 compute alpha for the
// 8 heads, broadcast via shuffle; all lanes stream the 128-float tail row and
// atomic-add into agg[head] (L2 resident).
// ---------------------------------------------------------------------------
__global__ void path_aggregate8(const int* __restrict__ pi, int P,
                                const float* __restrict__ s,
                                const float* __restrict__ mbuf,
                                const float* __restrict__ ssum,
                                const float* __restrict__ h1,
                                float* __restrict__ agg, int N) {
  int gw = (blockIdx.x * blockDim.x + threadIdx.x) >> 5;
  if (gw >= P) return;
  int lane = threadIdx.x & 31;
  int hd = pi[gw], md = pi[P + gw], tl = pi[2 * P + gw];
  size_t NH = (size_t)N * 8;
  const float* hrow = h1 + (size_t)tl * 128;
  __builtin_prefetch(hrow, 0, 1);  // global_prefetch_b8
  float alpha = 0.f;
  if (lane < 8) {
    float e = s[(size_t)hd * 8 + lane] + s[NH + (size_t)md * 8 + lane] +
              s[2 * NH + (size_t)tl * 8 + lane];
    e = (e > 0.f) ? e : NEG_SLOPE * e;
    float ex = expf(e - mbuf[(size_t)hd * 8 + lane]);
    alpha = ex / (ssum[(size_t)hd * 8 + lane] + EPS_F);
  }
  float* orow = agg + (size_t)hd * 128;
#pragma unroll
  for (int j = 0; j < 4; ++j) {
    int idx = lane + 32 * j;
    float a = __shfl(alpha, idx >> 4, 32);
    atomicAdd(&orow[idx], a * hrow[idx]);
  }
}

// Layer-2 aggregation (H=1, C=16): thread per (path, channel)
__global__ void path_aggregate1(const int* __restrict__ pi, int P,
                                const float* __restrict__ s,
                                const float* __restrict__ m2,
                                const float* __restrict__ ss2,
                                const float* __restrict__ h2,
                                float* __restrict__ agg, int N) {
  int t = blockIdx.x * blockDim.x + threadIdx.x;
  if (t >= P * 16) return;
  int p = t >> 4, c = t & 15;
  int hd = pi[p], md = pi[P + p], tl = pi[2 * P + p];
  float e = s[hd] + s[(size_t)N + md] + s[2 * (size_t)N + tl];
  e = (e > 0.f) ? e : NEG_SLOPE * e;
  float alpha = expf(e - m2[hd]) / (ss2[hd] + EPS_F);
  atomicAdd(&agg[(size_t)hd * 16 + c], alpha * h2[(size_t)tl * 16 + c]);
}

// agg1 + b1 -> ELU -> features for layer 2 (written in place of h1 buffer)
__global__ void elu_bias(const float* __restrict__ agg,
                         const float* __restrict__ b, float* __restrict__ o,
                         size_t total) {
  size_t i = (size_t)blockIdx.x * blockDim.x + threadIdx.x;
  if (i >= total) return;
  float v = agg[i] + b[i & 127];
  o[i] = (v > 0.f) ? v : (expf(v) - 1.f);
}

// agg2 + b2 -> log_softmax over 16 classes
__global__ void final_logsoftmax(const float* __restrict__ agg2,
                                 const float* __restrict__ b2,
                                 float* __restrict__ out, int N) {
  int n = blockIdx.x * blockDim.x + threadIdx.x;
  if (n >= N) return;
  float v[16];
  float m = -3.4e38f;
#pragma unroll
  for (int c = 0; c < 16; ++c) {
    v[c] = agg2[(size_t)n * 16 + c] + b2[c];
    m = fmaxf(m, v[c]);
  }
  float ssum = 0.f;
#pragma unroll
  for (int c = 0; c < 16; ++c) ssum += expf(v[c] - m);
  float l = logf(ssum);
#pragma unroll
  for (int c = 0; c < 16; ++c) out[(size_t)n * 16 + c] = v[c] - m - l;
}

// ---------------------------------------------------------------------------
// Host orchestration
// ---------------------------------------------------------------------------
extern "C" void kernel_launch(void* const* d_in, const int* in_sizes, int n_in,
                              void* d_out, int out_size, void* d_ws,
                              size_t ws_size, hipStream_t stream) {
  const float* x  = (const float*)d_in[0];
  const int*   pi = (const int*)d_in[1];
  const float* W1 = (const float*)d_in[2];
  const float* a1 = (const float*)d_in[3];
  const float* b1 = (const float*)d_in[4];
  const float* W2 = (const float*)d_in[5];
  const float* a2 = (const float*)d_in[6];
  const float* b2 = (const float*)d_in[7];
  float* out = (float*)d_out;

  const int FIN = 500, HC = 128, CLS = 16;
  const int N = in_sizes[0] / FIN;
  const int P = in_sizes[1] / 3;

  // workspace carve-out (256B aligned sections)
  char* ws = (char*)d_ws;
  size_t off = 0;
  auto alloc = [&](size_t bytes) -> void* {
    void* p = ws + off;
    off = (off + bytes + 255) & ~(size_t)255;
    return p;
  };
  __bf16* wb1  = (__bf16*)alloc((size_t)16 * 8 * 32 * 16 * 2);  // packed W1
  __bf16* wb2  = (__bf16*)alloc((size_t)4 * 1 * 32 * 16 * 2);   // packed W2
  float* h1    = (float*)alloc((size_t)N * HC * 4);   // GEMM1 out, later ELU feats
  float* s1    = (float*)alloc((size_t)3 * N * 8 * 4);
  float* m1    = (float*)alloc((size_t)N * 8 * 4);
  float* ss1   = (float*)alloc((size_t)N * 8 * 4);
  float* agg1  = (float*)alloc((size_t)N * HC * 4);
  float* h2    = (float*)alloc((size_t)N * CLS * 4);
  float* s2    = (float*)alloc((size_t)3 * N * 4);
  float* m2    = (float*)alloc((size_t)N * 4);
  float* ss2   = (float*)alloc((size_t)N * 4);
  float* agg2  = (float*)alloc((size_t)N * CLS * 4);

  const float NEG_INF = -__builtin_inff();
  auto cdiv = [](long a, long b) { return (unsigned)((a + b - 1) / b); };

  // ---- layer 1 ----
  pack_w_bf16<<<cdiv(16 * 8 * 32 * 16, 256), 256, 0, stream>>>(W1, 128, FIN, 16, 8, wb1);
  pack_w_bf16<<<cdiv(4 * 1 * 32 * 16, 256), 256, 0, stream>>>(W2, 16, 128, 4, 1, wb2);

  {  // GEMM1: h1 = x @ W1  (bf16 WMMA, f32 acc, async-LDS B pipeline)
    long waves = (N + 15) / 16;
    wmma_gemm_bf16<8, 500, 16, 500><<<cdiv(waves * 32, 256), 256, 0, stream>>>(
        x, (const unsigned int*)wb1, h1, HC, N);
  }

  node_logits8<<<cdiv((long)N * 8, 256), 256, 0, stream>>>(h1, a1, s1, N);

  fill_f32<<<2048, 256, 0, stream>>>(m1, NEG_INF, (size_t)N * 8);
  fill_f32<<<2048, 256, 0, stream>>>(ss1, 0.f, (size_t)N * 8);
  fill_f32<<<2048, 256, 0, stream>>>(agg1, 0.f, (size_t)N * HC);

  path_max<8><<<cdiv((long)P * 8, 256), 256, 0, stream>>>(pi, P, s1, N, m1);
  path_sumexp<8><<<cdiv((long)P * 8, 256), 256, 0, stream>>>(pi, P, s1, N, m1, ss1);
  path_aggregate8<<<cdiv((long)P * 32, 256), 256, 0, stream>>>(pi, P, s1, m1, ss1, h1, agg1, N);

  elu_bias<<<cdiv((long)N * HC, 256), 256, 0, stream>>>(agg1, b1, h1, (size_t)N * HC);

  // ---- layer 2 ----
  {  // GEMM2: h2 = feats @ W2
    long waves = (N + 15) / 16;
    wmma_gemm_bf16<1, 128, 4, 128><<<cdiv(waves * 32, 256), 256, 0, stream>>>(
        h1, (const unsigned int*)wb2, h2, CLS, N);
  }

  node_logits1<<<cdiv(N, 256), 256, 0, stream>>>(h2, a2, s2, N);

  fill_f32<<<2048, 256, 0, stream>>>(m2, NEG_INF, (size_t)N);
  fill_f32<<<2048, 256, 0, stream>>>(ss2, 0.f, (size_t)N);
  fill_f32<<<2048, 256, 0, stream>>>(agg2, 0.f, (size_t)N * CLS);

  path_max<1><<<cdiv(P, 256), 256, 0, stream>>>(pi, P, s2, N, m2);
  path_sumexp<1><<<cdiv(P, 256), 256, 0, stream>>>(pi, P, s2, N, m2, ss2);
  path_aggregate1<<<cdiv((long)P * 16, 256), 256, 0, stream>>>(pi, P, s2, m2, ss2, h2, agg2, N);

  final_logsoftmax<<<cdiv(N, 256), 256, 0, stream>>>(agg2, b2, out, N);
}
